// TopoGraphLayer_57930518888717
// MI455X (gfx1250) — matrix operand order repaired and checked by reading, not compiled, not run on hardware
//
#include <hip/hip_runtime.h>

#define TPB  256
#define NJ   16
#define NW   2
#define NT   2
#define DD   128
#define HH   128
#define NOUT (NJ + NW + NT)   // 20
#define LDSW 136              // padded bf16 row stride (elements)
#define CHUNK_ELEMS (128 * LDSW)            // one staged weight chunk (bf16 elems)
#define CHUNK_BYTES (CHUNK_ELEMS * 2)       // 34816 bytes
#define N_CHUNKS 34                         // 7*3 edge + 5 jet + 4 + 4 wt

typedef __attribute__((ext_vector_type(16))) __bf16 bf16x16;
typedef __attribute__((ext_vector_type(8)))  __bf16 bf16x8;
typedef __attribute__((ext_vector_type(8)))  float  f32x8;

__device__ __forceinline__ unsigned short f2bf(float f) {
  unsigned int u = __float_as_uint(f);
  u += 0x7FFFu + ((u >> 16) & 1u);       // round-to-nearest-even
  return (unsigned short)(u >> 16);
}
__device__ __forceinline__ unsigned pk2(float x, float y) {
  return (unsigned)f2bf(x) | ((unsigned)f2bf(y) << 16);
}

struct Smem {
  unsigned short* sJ;    // 16x136 bf16 jets
  unsigned short* sW;    // 16x136 bf16 nodes_w (padded)
  unsigned short* sT;    // 16x136 bf16 nodes_top (padded)
  unsigned short* sA;    // 16x136 bf16 scratch A operand
  unsigned short* sWt;   // 128x136 bf16 weight chunk (async-DMA destination)
  unsigned short* h1;    // 256x136 bf16 edge hidden
  float* projA;          // 16x128
  float* projC;          // 16x128
  float* pool0;          // 16x128
  float* pool1;          // 16x128
  float* pool2;          // 16x128
  float* sB1;            // 128
  float* sB2;            // 128
  float* dmJ;            // 16
  float* dmW;            // 16
  float* dmT;            // 16
  float* misc;           // [0] = 1/max(sum(mask),1)
};

// ---- async DMA one pre-transposed bf16 weight chunk (global -> LDS), ASYNCcnt ----
__device__ __forceinline__ void async_chunk(unsigned short* sWt,
                                            const unsigned short* src) {
  unsigned lds = (unsigned)(uintptr_t)sWt + (unsigned)(threadIdx.x * 16u);
  unsigned long long g =
      (unsigned long long)(uintptr_t)((const char*)src) + threadIdx.x * 16u;
  // 34816 B = 8 full 4096B sweeps + half sweep
#pragma unroll
  for (int it = 0; it < 8; ++it) {
    unsigned l = lds + (unsigned)(it * 4096);
    unsigned long long ga = g + (unsigned long long)(it * 4096);
    asm volatile("global_load_async_to_lds_b128 %0, %1, off"
                 :: "v"(l), "v"(ga) : "memory");
  }
  if (threadIdx.x < 128) {
    unsigned l = lds + 32768u;
    unsigned long long ga = g + 32768ull;
    asm volatile("global_load_async_to_lds_b128 %0, %1, off"
                 :: "v"(l), "v"(ga) : "memory");
  }
}
__device__ __forceinline__ void async_wait() {
  asm volatile("s_wait_asynccnt 0x0" ::: "memory");
}

// ---- one 16x16 f32 tile of A(16xK bf16, LDS) @ B(KxN via transposed LDS), K=128 ----
__device__ __forceinline__ void wmma_k128(const unsigned short* sAa, int lda,
                                          const unsigned short* sBt, int ntile,
                                          f32x8& acc, int lane)
{
  const int half = lane >> 4;
  const int mr   = lane & 15;
  const int nr   = (ntile << 4) + (lane & 15);
  const unsigned short* pa0 = sAa + mr * lda + half * 8;
  const unsigned short* pb0 = sBt + nr * LDSW + half * 8;
#pragma unroll
  for (int kc = 0; kc < 128; kc += 32) {
    bf16x8 alo = *(const bf16x8*)(pa0 + kc);
    bf16x8 ahi = *(const bf16x8*)(pa0 + kc + 16);
    bf16x8 blo = *(const bf16x8*)(pb0 + kc);
    bf16x8 bhi = *(const bf16x8*)(pb0 + kc + 16);
    bf16x16 av, bv;
#pragma unroll
    for (int e = 0; e < 8; ++e) {
      av[e] = alo[e]; av[8 + e] = ahi[e];
      bv[e] = blo[e]; bv[8 + e] = bhi[e];
    }
    acc = __builtin_amdgcn_wmma_f32_16x16x32_bf16(false, av, false, bv,
                                                  (short)0, acc, false, false);
  }
}

__device__ __forceinline__ void stage_feat(unsigned short* s, const float* __restrict__ x,
                                           int nvalid) {
  for (int q = threadIdx.x; q < 16 * 32; q += TPB) {     // 4 cols per iter
    int r = q >> 5, c4 = (q & 31) << 2;
    float4 v = make_float4(0.f, 0.f, 0.f, 0.f);
    if (r < nvalid) v = *(const float4*)&x[r * 128 + c4];
    unsigned long long pk = ((unsigned long long)pk2(v.z, v.w) << 32) | pk2(v.x, v.y);
    *(unsigned long long*)&s[r * LDSW + c4] = pk;
  }
}

// ---------------- fused edge block: pooled = pool_j MLP2(concat(src_i, dst_j)) ----------
__device__ void edge_block(const Smem& S,
                           const unsigned short* srcS, const unsigned short* dstS,
                           const unsigned short* __restrict__ wsE,   // W1a,W1b,W2 chunks
                           const float* __restrict__ eb1, const float* __restrict__ eb2,
                           const float* dmask, float inv_count, float* pooled)
{
  const int tid  = threadIdx.x;
  const int lane = tid & 31;
  const int wave = tid >> 5;
  const int half = lane >> 4;

  async_chunk(S.sWt, wsE);                                 // W1 src part
  for (int i = tid; i < 128; i += TPB) { S.sB1[i] = eb1[i]; S.sB2[i] = eb2[i]; }
  async_wait();
  __syncthreads();
  {                                                        // projA = src @ W1_src
    f32x8 acc = {};
    wmma_k128(srcS, LDSW, S.sWt, wave, acc, lane);
    int n = (wave << 4) + (lane & 15);
#pragma unroll
    for (int r = 0; r < 8; ++r) S.projA[(half * 8 + r) * 128 + n] = acc[r];
  }
  __syncthreads();
  async_chunk(S.sWt, wsE + CHUNK_ELEMS);                   // W1 dst part
  async_wait();
  __syncthreads();
  {                                                        // projC = dst @ W1_dst
    f32x8 acc = {};
    wmma_k128(dstS, LDSW, S.sWt, wave, acc, lane);
    int n = (wave << 4) + (lane & 15);
#pragma unroll
    for (int r = 0; r < 8; ++r) S.projC[(half * 8 + r) * 128 + n] = acc[r];
  }
  __syncthreads();
  async_chunk(S.sWt, wsE + 2 * CHUNK_ELEMS);               // W2 (DMA overlaps h1 build)
  // h1[i*16+j][c] = relu(projA[i][c] + projC[j][c] + b1[c]) -> bf16, 4 cols/iter
  for (int q = tid; q < 256 * 32; q += TPB) {
    int row = q >> 5, c4 = (q & 31) << 2;
    int i = row >> 4, j = row & 15;
    float4 a  = *(const float4*)&S.projA[i * 128 + c4];
    float4 c  = *(const float4*)&S.projC[j * 128 + c4];
    float4 bb = *(const float4*)&S.sB1[c4];
    unsigned lo = pk2(fmaxf(a.x + c.x + bb.x, 0.f), fmaxf(a.y + c.y + bb.y, 0.f));
    unsigned hi = pk2(fmaxf(a.z + c.z + bb.z, 0.f), fmaxf(a.w + c.w + bb.w, 0.f));
    *(unsigned long long*)&S.h1[row * LDSW + c4] =
        ((unsigned long long)hi << 32) | lo;
  }
  async_wait();
  __syncthreads();
  // layer 2 + masked mean-pool; M-tile mt == source node i (rows are all dst j)
  for (int t = wave; t < 128; t += 8) {
    int mt = t >> 3, nt = t & 7;
    f32x8 acc = {};
    wmma_k128(S.h1 + mt * 16 * LDSW, LDSW, S.sWt, nt, acc, lane);
    int n = (nt << 4) + (lane & 15);
    float s = 0.f;
#pragma unroll
    for (int r = 0; r < 8; ++r) {
      int j = half * 8 + r;
      float v = fmaxf(acc[r] + S.sB2[n], 0.f);
      s += v * dmask[j];
    }
    s += __shfl_xor(s, 16, 32);
    if (half == 0) pooled[mt * 128 + n] = s * inv_count;
  }
  __syncthreads();
}

// ---------------- node MLP2 over concatenated chunks (K = 128 * nchunks) ----------------
__device__ void node_net(const Smem& S, const unsigned short* feat,
                         const float* p0, const float* p1, const float* p2,
                         const unsigned short* __restrict__ wsN, int nchunks,
                         const float* __restrict__ b1, const float* __restrict__ b2,
                         int nvalid, const float* rowScale, float* __restrict__ outRow)
{
  const int tid  = threadIdx.x;
  const int lane = tid & 31;
  const int wave = tid >> 5;
  const int half = lane >> 4;
  const float* pools[3] = { p0, p1, p2 };

  for (int i = tid; i < 128; i += TPB) { S.sB1[i] = b1[i]; S.sB2[i] = b2[i]; }

  f32x8 acc = {};
  for (int c = 0; c < nchunks; ++c) {
    async_chunk(S.sWt, wsN + (size_t)c * CHUNK_ELEMS);     // DMA overlaps conversion
    const unsigned short* Achunk = feat;
    if (c > 0) {
      const float* p = pools[c - 1];
      for (int q = tid; q < 16 * 32; q += TPB) {
        int r = q >> 5, c4 = (q & 31) << 2;
        float4 v = *(const float4*)&p[r * 128 + c4];
        *(unsigned long long*)&S.sA[r * LDSW + c4] =
            ((unsigned long long)pk2(v.z, v.w) << 32) | pk2(v.x, v.y);
      }
      Achunk = S.sA;
    }
    async_wait();
    __syncthreads();
    wmma_k128(Achunk, LDSW, S.sWt, wave, acc, lane);
    __syncthreads();
  }
  {   // hidden = relu(acc + b1) -> sA (bf16), becomes layer-2 A operand
    int n = (wave << 4) + (lane & 15);
#pragma unroll
    for (int r = 0; r < 8; ++r) {
      float v = fmaxf(acc[r] + S.sB1[n], 0.f);
      S.sA[(half * 8 + r) * LDSW + n] = f2bf(v);
    }
  }
  async_chunk(S.sWt, wsN + (size_t)nchunks * CHUNK_ELEMS); // W2
  async_wait();
  __syncthreads();
  f32x8 acc2 = {};
  wmma_k128(S.sA, LDSW, S.sWt, wave, acc2, lane);
  {
    int n = (wave << 4) + (lane & 15);
#pragma unroll
    for (int r = 0; r < 8; ++r) {
      int m = half * 8 + r;
      float v = fmaxf(acc2[r] + S.sB2[n], 0.f);
      if (rowScale) v *= rowScale[m];
      if (m < nvalid) outRow[m * 128 + n] = v;
    }
  }
  __syncthreads();
}

// ---- one-shot prep: convert + transpose every 128x128 weight chunk to bf16 in ws ------
__global__ void __launch_bounds__(TPB)
prep_weights(const float* __restrict__ eW1, const float* __restrict__ eW2,
             const float* __restrict__ jW1, const float* __restrict__ jW2,
             const float* __restrict__ wtW1, const float* __restrict__ wtW2,
             unsigned short* __restrict__ ws)
{
  const int c = blockIdx.x;
  const float* W;
  if (c < 21) {                       // edge nets: [W1a, W1b, W2] per net
    int e = c / 3, s = c % 3;
    W = (s < 2) ? (eW1 + (size_t)e * 256 * 128 + (size_t)s * 128 * 128)
                : (eW2 + (size_t)e * 128 * 128);
  } else if (c < 26) {                // jet net: 4x W1 chunks + W2
    int s = c - 21;
    W = (s < 4) ? (jW1 + (size_t)s * 128 * 128) : jW2;
  } else {                            // wt nets: 3x W1 chunks + W2, twice
    int g = (c - 26) / 4, s = (c - 26) % 4;
    W = (s < 3) ? (wtW1 + (size_t)g * 384 * 128 + (size_t)s * 128 * 128)
                : (wtW2 + (size_t)g * 128 * 128);
  }
  unsigned short* dst = ws + (size_t)c * CHUNK_ELEMS;
  for (int idx = threadIdx.x; idx < 128 * 128; idx += TPB) {
    int k = idx >> 7, n = idx & 127;
    dst[n * LDSW + k] = f2bf(W[idx]);          // idx == k*128 + n
  }
}

// ----------------------------------- main kernel ---------------------------------------
__global__ void __launch_bounds__(TPB, 1)
topo_graph_kernel(const float* __restrict__ jets, const float* __restrict__ nw,
                  const float* __restrict__ ntop, const int* __restrict__ mask,
                  const unsigned short* __restrict__ ws,
                  const float* __restrict__ eb1, const float* __restrict__ eb2,
                  const float* __restrict__ jb1, const float* __restrict__ jb2,
                  const float* __restrict__ wtb1, const float* __restrict__ wtb2,
                  float* __restrict__ out)
{
  extern __shared__ char smem[];
  Smem S;
  char* p = smem;
  S.sJ    = (unsigned short*)p; p += 16 * LDSW * 2;
  S.sW    = (unsigned short*)p; p += 16 * LDSW * 2;
  S.sT    = (unsigned short*)p; p += 16 * LDSW * 2;
  S.sA    = (unsigned short*)p; p += 16 * LDSW * 2;
  S.sWt   = (unsigned short*)p; p += 128 * LDSW * 2;
  S.h1    = (unsigned short*)p; p += 256 * LDSW * 2;
  S.projA = (float*)p; p += 16 * 128 * 4;
  S.projC = (float*)p; p += 16 * 128 * 4;
  S.pool0 = (float*)p; p += 16 * 128 * 4;
  S.pool1 = (float*)p; p += 16 * 128 * 4;
  S.pool2 = (float*)p; p += 16 * 128 * 4;
  S.sB1   = (float*)p; p += 128 * 4;
  S.sB2   = (float*)p; p += 128 * 4;
  S.dmJ   = (float*)p; p += 16 * 4;
  S.dmW   = (float*)p; p += 16 * 4;
  S.dmT   = (float*)p; p += 16 * 4;
  S.misc  = (float*)p;

  const int b   = blockIdx.x;
  const int tid = threadIdx.x;

  stage_feat(S.sJ, jets + (size_t)b * NJ * DD, NJ);
  stage_feat(S.sW, nw   + (size_t)b * NW * DD, NW);
  stage_feat(S.sT, ntop + (size_t)b * NT * DD, NT);
  if (tid < 16) {
    S.dmJ[tid] = mask[b * NJ + tid] ? 1.f : 0.f;
    S.dmW[tid] = (tid < NW) ? 1.f : 0.f;
    S.dmT[tid] = (tid < NT) ? 1.f : 0.f;
  }
  __syncthreads();
  if (tid == 0) {
    float c = 0.f;
    for (int i = 0; i < 16; ++i) c += S.dmJ[i];
    S.misc[0] = 1.f / fmaxf(c, 1.f);
  }
  __syncthreads();
  const float invJ = S.misc[0];
  float* outB = out + (size_t)b * NOUT * HH;

  // ---- jet update ----
  edge_block(S, S.sJ, S.sJ, ws + 0 * 3 * CHUNK_ELEMS, eb1 + 0 * 128, eb2 + 0 * 128,
             S.dmJ, invJ, S.pool0);                                   // p_jj
  edge_block(S, S.sJ, S.sW, ws + 1 * 3 * CHUNK_ELEMS, eb1 + 1 * 128, eb2 + 1 * 128,
             S.dmW, 0.5f, S.pool1);                                   // p_jw
  edge_block(S, S.sJ, S.sT, ws + 2 * 3 * CHUNK_ELEMS, eb1 + 2 * 128, eb2 + 2 * 128,
             S.dmT, 0.5f, S.pool2);                                   // p_jt
  node_net(S, S.sJ, S.pool0, S.pool1, S.pool2, ws + 21 * CHUNK_ELEMS, 4,
           jb1, jb2, NJ, S.dmJ, outB);

  // ---- W update ----
  edge_block(S, S.sW, S.sJ, ws + 3 * 3 * CHUNK_ELEMS, eb1 + 3 * 128, eb2 + 3 * 128,
             S.dmJ, invJ, S.pool0);                                   // p_wj
  edge_block(S, S.sW, S.sT, ws + 4 * 3 * CHUNK_ELEMS, eb1 + 4 * 128, eb2 + 4 * 128,
             S.dmT, 0.5f, S.pool1);                                   // p_wt
  node_net(S, S.sW, S.pool0, S.pool1, nullptr, ws + 26 * CHUNK_ELEMS, 3,
           wtb1 + 0 * 128, wtb2 + 0 * 128, NW, nullptr, outB + NJ * HH);

  // ---- top update ----
  edge_block(S, S.sT, S.sJ, ws + 5 * 3 * CHUNK_ELEMS, eb1 + 5 * 128, eb2 + 5 * 128,
             S.dmJ, invJ, S.pool0);                                   // p_tj
  edge_block(S, S.sT, S.sW, ws + 6 * 3 * CHUNK_ELEMS, eb1 + 6 * 128, eb2 + 6 * 128,
             S.dmW, 0.5f, S.pool1);                                   // p_tw
  node_net(S, S.sT, S.pool0, S.pool1, nullptr, ws + 30 * CHUNK_ELEMS, 3,
           wtb1 + 1 * 128, wtb2 + 1 * 128, NT, nullptr, outB + (NJ + NW) * HH);
}

extern "C" void kernel_launch(void* const* d_in, const int* in_sizes, int n_in,
                              void* d_out, int out_size, void* d_ws, size_t ws_size,
                              hipStream_t stream) {
  const float* jets = (const float*)d_in[0];
  const float* nw   = (const float*)d_in[1];
  const float* ntop = (const float*)d_in[2];
  const int*   mask = (const int*)d_in[3];
  const float* eW1  = (const float*)d_in[4];
  const float* eb1  = (const float*)d_in[5];
  const float* eW2  = (const float*)d_in[6];
  const float* eb2  = (const float*)d_in[7];
  const float* jW1  = (const float*)d_in[8];
  const float* jb1  = (const float*)d_in[9];
  const float* jW2  = (const float*)d_in[10];
  const float* jb2  = (const float*)d_in[11];
  const float* wtW1 = (const float*)d_in[12];
  const float* wtb1 = (const float*)d_in[13];
  const float* wtW2 = (const float*)d_in[14];
  const float* wtb2 = (const float*)d_in[15];

  unsigned short* ws = (unsigned short*)d_ws;   // needs N_CHUNKS*CHUNK_BYTES ~ 1.13 MB

  const int B = in_sizes[0] / (NJ * DD);        // 1024

  prep_weights<<<N_CHUNKS, TPB, 0, stream>>>(eW1, eW2, jW1, jW2, wtW1, wtW2, ws);

  size_t smem = 0;
  smem += 4u * (16 * LDSW * 2);       // sJ sW sT sA
  smem += 128 * LDSW * 2;             // sWt
  smem += 256 * LDSW * 2;             // h1
  smem += 5u * (16 * 128 * 4);        // projA projC pool0..2
  smem += 2u * (128 * 4);             // sB1 sB2
  smem += 3u * (16 * 4);              // dmJ dmW dmT
  smem += 16;                         // misc

  topo_graph_kernel<<<B, TPB, smem, stream>>>(
      jets, nw, ntop, mask, ws, eb1, eb2, jb1, jb2, wtb1, wtb2, (float*)d_out);
}
